// SparseMoE_515396076109
// MI455X (gfx1250) — matrix operand order repaired and checked by reading, not compiled, not run on hardware
//
#include <hip/hip_runtime.h>

// ---------------- types ----------------
typedef unsigned short u16;
typedef unsigned int   u32;
typedef __attribute__((ext_vector_type(16))) __bf16 v16bf;
typedef __attribute__((ext_vector_type(8)))  float  v8f;
typedef int v4i __attribute__((vector_size(16)));   // matches builtin's param pointee

#define T_TOK 8192   // B*S
#define DDIM  768
#define EEXP  8
#define HDIM  3072

// CDNA5 async global->LDS path (guarded: falls back to ds stores if absent)
#if __has_builtin(__builtin_amdgcn_global_load_async_to_lds_b128) && \
    __has_builtin(__builtin_amdgcn_s_wait_asynccnt)
#define USE_ASYNC_LDS 1
#endif

typedef __attribute__((address_space(1))) v4i gv4i;   // global (prints as __device__)
typedef __attribute__((address_space(3))) v4i lv4i;   // LDS    (prints as __shared__)

__device__ __forceinline__ void copy16_to_lds(const void* g, void* l) {
#ifdef USE_ASYNC_LDS
  __builtin_amdgcn_global_load_async_to_lds_b128((gv4i*)g, (lv4i*)l, 0, 0);
#else
  *(uint4*)l = *(const uint4*)g;
#endif
}

__device__ __forceinline__ u16 f2bf(float f) {
  u32 u = __float_as_uint(f);
  u32 r = u + 0x7FFFu + ((u >> 16) & 1u);   // round-to-nearest-even
  return (u16)(r >> 16);
}

// ---------------- fp32 -> bf16 bulk convert (activations) ----------------
__global__ void cvt_bf16_kernel(const float* __restrict__ x, u16* __restrict__ xb, int n) {
  int i = blockIdx.x * blockDim.x + threadIdx.x;
  int stride = gridDim.x * blockDim.x;
  for (; i < n; i += stride) xb[i] = f2bf(x[i]);
}

// ---------------- weight pre-swizzle: f32 [E,K,N] -> bf16 frag-major blocks ----
// Output layout per expert: block (j=n>>4, kb=k>>4) at (j*(K/16)+kb)*256,
// interior (n&15)*16 + (k&15). A 32K x 16N WMMA B-fragment tile is then two
// contiguous 512-byte runs -> pure b128 async copies into LDS at GEMM time.
__global__ void swizzle_w_kernel(const float* __restrict__ W, u16* __restrict__ Wb,
                                 int K, int N) {
  const int nb_per_mat = (K >> 4) * (N >> 4);
  const int gid = blockIdx.x * 8 + (threadIdx.x >> 5);   // global 16x16 block id
  const int sub = threadIdx.x & 31;
  const int mat = gid / nb_per_mat;                      // expert
  const int rem = gid % nb_per_mat;
  const int j   = rem / (K >> 4);                        // n-block
  const int kb  = rem % (K >> 4);                        // k-block
  const float* src = W + (size_t)mat * K * N;
  u16* dst = Wb + (size_t)mat * K * N + (size_t)rem * 256;

  const int l  = sub & 15;            // n & 15
  const int kh = (sub >> 4) << 3;     // 0 or 8
  const int n  = (j << 4) + l;
  u16 tmp[8];
#pragma unroll
  for (int i = 0; i < 8; i++) {
    int k = (kb << 4) + kh + i;
    tmp[i] = f2bf(src[(size_t)k * N + n]);
  }
  *(uint4*)(dst + l * 16 + kh) = *(uint4*)tmp;
}

// ---------------- noisy top-2 router ----------------
__global__ void router_kernel(const float* __restrict__ x,
                              const float* __restrict__ noise,
                              const float* __restrict__ w_route,
                              const float* __restrict__ b_route,
                              const float* __restrict__ w_noise,
                              const float* __restrict__ b_noise,
                              float* __restrict__ gates) {
  int t = blockIdx.x * blockDim.x + threadIdx.x;
  if (t >= T_TOK) return;

  float lg[EEXP], nl[EEXP];
#pragma unroll
  for (int e = 0; e < EEXP; e++) { lg[e] = b_route[e]; nl[e] = b_noise[e]; }

  const float* xr = x + (size_t)t * DDIM;
  for (int d = 0; d < DDIM; d++) {
    float xv = xr[d];
#pragma unroll
    for (int e = 0; e < EEXP; e++) {
      lg[e] = fmaf(xv, w_route[d * EEXP + e], lg[e]);
      nl[e] = fmaf(xv, w_noise[d * EEXP + e], nl[e]);
    }
  }

  float noisy[EEXP];
#pragma unroll
  for (int e = 0; e < EEXP; e++) {
    float v  = nl[e];
    float sp = (v > 20.f) ? v : log1pf(__expf(v));   // softplus
    noisy[e] = lg[e] + noise[(size_t)t * EEXP + e] * sp;
  }

  int i1 = 0;
#pragma unroll
  for (int e = 1; e < EEXP; e++) if (noisy[e] > noisy[i1]) i1 = e;
  int i2 = (i1 == 0) ? 1 : 0;
#pragma unroll
  for (int e = 0; e < EEXP; e++) {
    if (e == i1) continue;
    if (noisy[e] > noisy[i2]) i2 = e;
  }

  float v1 = noisy[i1], v2 = noisy[i2];
  float m  = fmaxf(v1, v2);
  float e1 = __expf(v1 - m), e2 = __expf(v2 - m);
  float s  = e1 + e2;

#pragma unroll
  for (int e = 0; e < EEXP; e++) gates[(size_t)t * EEXP + e] = 0.f;
  gates[(size_t)t * EEXP + i1] = e1 / s;
  gates[(size_t)t * EEXP + i2] = e2 / s;
}

// ---------------- tiled bf16 WMMA GEMM (double-buffered async pipeline) ----------
// MODE 0: outB = relu(A@W + bias)            (bf16 store)
// MODE 1: outF (+)= gate * (A@W + bias)      (fp32; expert 0 writes, else accumulates)
// Block: 256 threads = 8 waves. Macro-tile 128(M) x 128(N), K-step 32.
// Wave (mw = wid&3, nw = wid>>2) owns a 32x64 sub-tile: 2 A-frags x 4 B-frags
// -> 8 back-to-back v_wmma_f32_16x16x32_bf16 per K-step.
// Tile k+1 is async-staged into the ping-pong LDS buffer during compute on tile k;
// s_wait_asynccnt(3) retires only the older load group (ASYNCcnt is in-order).
template <int MODE>
__global__ __launch_bounds__(256) void moe_gemm_kernel(
    const u16*  __restrict__ A,     // [T, Ktot] bf16 row-major
    const u16*  __restrict__ Wb,    // frag-major swizzled bf16 weights
    const float* __restrict__ bias, // [Ntot]
    u16*   __restrict__ outB,       // MODE 0: [T, Ntot] bf16
    float* __restrict__ outF,       // MODE 1: [T, Ntot] f32
    const float* __restrict__ gates,// [T, EEXP]
    int Ktot, int Ntot, int expert)
{
  __shared__ __align__(16) u16 ldsA[2][128 * 32];    // row-major M x K tiles (2 x 8 KB)
  __shared__ __align__(16) u16 ldsB[2][8 * 32 * 16]; // frag-ordered tiles    (2 x 8 KB)

  const int tid  = threadIdx.x;
  const int lane = tid & 31;
  const int wid  = tid >> 5;
  const int mw   = wid & 3;       // M-wave 0..3
  const int nw   = wid >> 2;      // N-wave 0..1
  const int t0   = blockIdx.x * 128;
  const int n0   = blockIdx.y * 128;
  const int KB   = Ktot >> 4;     // k-blocks in swizzled weights

  // per-thread staging coordinates (invariant across K-steps)
  const int arow0 = tid >> 2;              // A rows for copies 0 / 1
  const int acp   = (tid & 3) << 3;        // A col offset (bf16 elems)
  const int bunit = tid >> 4;              // B 512B block 0..15 = (j, kbrel)
  const int binn  = (tid & 15) << 3;       // B inner offset (u16 elems)
  const size_t bsrc_base =
      ((size_t)((n0 >> 4) + (bunit >> 1)) * KB + (bunit & 1)) * 256 + binn;

  v8f acc[2][4] = {};

  const int ksteps = Ktot >> 5;

  // stage K-tile `ks` into buffer `b`: 3 async b128 copies per thread
  auto stage = [&](int ks, int b) {
    const int k0 = ks << 5;
    copy16_to_lds(A + (size_t)(t0 + arow0) * Ktot + k0 + acp,
                  &ldsA[b][(arow0 << 5) + acp]);
    copy16_to_lds(A + (size_t)(t0 + arow0 + 64) * Ktot + k0 + acp,
                  &ldsA[b][((arow0 + 64) << 5) + acp]);
    copy16_to_lds(Wb + bsrc_base + (size_t)(ks << 1) * 256,
                  &ldsB[b][(bunit << 8) + binn]);
  };

  stage(0, 0);

  for (int ks = 0; ks < ksteps; ks++) {
    const int buf = ks & 1;
    if (ks + 1 < ksteps) {
      stage(ks + 1, buf ^ 1);            // prefetch next tile behind compute
#ifdef USE_ASYNC_LDS
      __builtin_amdgcn_s_wait_asynccnt(3);   // retire only tile-ks loads
#endif
    } else {
#ifdef USE_ASYNC_LDS
      __builtin_amdgcn_s_wait_asynccnt(0);
#endif
    }
    __syncthreads();

    // --- load all fragments, then 8 WMMAs back-to-back ---
    {
      const int mlane = lane & 15;
      const int kh    = (lane >> 4) << 4;    // 0 or 16 (A frag K-half per lane)
      v16bf afr[2], bfr[4];
#pragma unroll
      for (int mi = 0; mi < 2; mi++) {
        const u16* pa = &ldsA[buf][(((mw << 5) + (mi << 4) + mlane) << 5) + kh];
        ((uint4*)&afr[mi])[0] = *(const uint4*)(pa);        // K kh..kh+7
        ((uint4*)&afr[mi])[1] = *(const uint4*)(pa + 8);    // K kh+8..kh+15
      }
#pragma unroll
      for (int j = 0; j < 4; j++) {
        const u16* pb = &ldsB[buf][((((nw << 2) + j) << 5) + lane) << 4];
        ((uint4*)&bfr[j])[0] = *(const uint4*)(pb);
        ((uint4*)&bfr[j])[1] = *(const uint4*)(pb + 8);
      }
#pragma unroll
      for (int mi = 0; mi < 2; mi++)
#pragma unroll
        for (int j = 0; j < 4; j++)
          acc[mi][j] = __builtin_amdgcn_wmma_f32_16x16x32_bf16(
              false, afr[mi], false, bfr[j], (short)0, acc[mi][j], false, false);
    }
    __syncthreads();   // buf may be overwritten by stage() next iteration
  }

  // --- epilogue: C layout VGPR r -> M = r + 8*(lane>>4), N = lane&15 ---
  const int mb0 = t0 + (mw << 5) + ((lane >> 4) << 3);
  const int nb0 = n0 + (nw << 6) + (lane & 15);
#pragma unroll
  for (int mi = 0; mi < 2; mi++) {
#pragma unroll
    for (int r = 0; r < 8; r++) {
      const int m = mb0 + (mi << 4) + r;
      float g = 0.f;
      if (MODE == 1) g = gates[(size_t)m * EEXP + expert];
#pragma unroll
      for (int j = 0; j < 4; j++) {
        const int n = nb0 + (j << 4);
        float v = acc[mi][j][r] + bias[n];
        if (MODE == 0) {
          v = fmaxf(v, 0.f);
          outB[(size_t)m * Ntot + n] = f2bf(v);
        } else {
          size_t idx = (size_t)m * Ntot + n;
          if (expert == 0)   outF[idx] = g * v;       // initializes d_out
          else if (g != 0.f) outF[idx] += g * v;      // top-2 sparse accumulate
        }
      }
    }
  }
}

// ---------------- launch ----------------
extern "C" void kernel_launch(void* const* d_in, const int* in_sizes, int n_in,
                              void* d_out, int out_size, void* d_ws, size_t ws_size,
                              hipStream_t stream) {
  const float* x       = (const float*)d_in[0];
  const float* noise   = (const float*)d_in[1];
  const float* w_route = (const float*)d_in[2];
  const float* b_route = (const float*)d_in[3];
  const float* w_noise = (const float*)d_in[4];
  const float* b_noise = (const float*)d_in[5];
  const float* W1      = (const float*)d_in[6];
  const float* b1      = (const float*)d_in[7];
  const float* W2      = (const float*)d_in[8];
  const float* b2      = (const float*)d_in[9];
  float* out = (float*)d_out;
  char*  ws  = (char*)d_ws;

  // workspace: gates 0.5MB | xb 12.6MB | hb 50.3MB | W1b 37.7MB | W2b 37.7MB
  const size_t szGates = 512 * 1024;
  const size_t szXb    = (size_t)T_TOK * DDIM * 2;
  const size_t szHb    = (size_t)T_TOK * HDIM * 2;
  const size_t szW     = (size_t)EEXP * DDIM * HDIM * 2;
  float* gates = (float*)ws;
  u16*   xb    = (u16*)(ws + szGates);
  u16*   hb    = (u16*)(ws + szGates + szXb);
  u16*   w1b   = (u16*)(ws + szGates + szXb + szHb);
  u16*   w2b   = (u16*)(ws + szGates + szXb + szHb + szW);

  cvt_bf16_kernel<<<512, 256, 0, stream>>>(x, xb, T_TOK * DDIM);

  const int nblk = (DDIM >> 4) * (HDIM >> 4);            // 9216 16x16 blocks/expert
  swizzle_w_kernel<<<EEXP * nblk / 8, 256, 0, stream>>>(W1, w1b, DDIM, HDIM);
  swizzle_w_kernel<<<EEXP * nblk / 8, 256, 0, stream>>>(W2, w2b, HDIM, DDIM);

  router_kernel<<<T_TOK / 256, 256, 0, stream>>>(x, noise, w_route, b_route,
                                                 w_noise, b_noise, gates);

  for (int e = 0; e < EEXP; e++) {
    // Hb = relu(Xb @ W1[e] + b1[e])          M=8192, N=3072, K=768
    moe_gemm_kernel<0><<<dim3(T_TOK / 128, HDIM / 128), 256, 0, stream>>>(
        xb, w1b + (size_t)e * DDIM * HDIM, b1 + (size_t)e * HDIM,
        hb, nullptr, nullptr, DDIM, HDIM, e);
    // Out (+)= gate_e * (Hb @ W2[e] + b2[e]) M=8192, N=768, K=3072
    moe_gemm_kernel<1><<<dim3(T_TOK / 128, DDIM / 128), 256, 0, stream>>>(
        hb, w2b + (size_t)e * HDIM * DDIM, b2 + (size_t)e * DDIM,
        nullptr, out, gates, HDIM, DDIM, e);
  }
}
